// MdTripletLoss_15874199126571
// MI455X (gfx1250) — matrix operand order repaired and checked by reading, not compiled, not run on hardware
//
#include <hip/hip_runtime.h>
#include <hip/hip_bf16.h>

#define EPS 1e-6f
#define BS   4096
#define DDIM 2048

typedef __attribute__((ext_vector_type(16))) __bf16 v16bf;
typedef __attribute__((ext_vector_type(8)))  __bf16 v8bf;
typedef __attribute__((ext_vector_type(8)))  float  v8f;

// ---------------------------------------------------------------------------
// Kernel 1: fp32 -> bf16 conversion; for h2 also compute c2[j] = sq2 - 2*eps*s2
// grid: 2*BS blocks of 256 threads. Blocks [0,BS) convert h1, [BS,2BS) do h2.
// ---------------------------------------------------------------------------
__global__ void __launch_bounds__(256) prep_kernel(
    const float* __restrict__ h1, const float* __restrict__ h2,
    __bf16* __restrict__ b1, __bf16* __restrict__ b2, float* __restrict__ c2) {
  int r = blockIdx.x;
  if (r < BS) {
    const float* src = h1 + (size_t)r * DDIM;
    __bf16* dst = b1 + (size_t)r * DDIM;
    for (int k = threadIdx.x; k < DDIM; k += 256)
      dst[k] = (__bf16)src[k];
  } else {
    int j = r - BS;
    const float* src = h2 + (size_t)j * DDIM;
    __bf16* dst = b2 + (size_t)j * DDIM;
    float sq = 0.0f, s = 0.0f;
    for (int k = threadIdx.x; k < DDIM; k += 256) {
      float v = src[k];
      dst[k] = (__bf16)v;
      sq += v * v;
      s  += v;
    }
    __shared__ float r0[256], r1[256];
    r0[threadIdx.x] = sq; r1[threadIdx.x] = s;
    __syncthreads();
    for (int off = 128; off > 0; off >>= 1) {
      if (threadIdx.x < off) {
        r0[threadIdx.x] += r0[threadIdx.x + off];
        r1[threadIdx.x] += r1[threadIdx.x + off];
      }
      __syncthreads();
    }
    if (threadIdx.x == 0) c2[j] = r0[0] - 2.0f * EPS * r1[0];
  }
}

// ---------------------------------------------------------------------------
// Kernel 2: WMMA bf16 GEMM producing score[i,j] = c2[j] - 2*dot(h1_i, h2_j).
// 128 threads = 4 waves in a 2x2 grid; WG tile 128(M) x 128(N); each wave
// computes a 64x64 sub-tile as 4x4 fragments of 16x16 with K stepped by 32
// (v_wmma_f32_16x16x32_bf16). 16 b128 loads feed 16 WMMAs per K-step
// (1.0 loads/WMMA vs 2.5 before), with operand reuse across the 4x4 block.
//
// A fragment (16x32 bf16, M=lane&15): lanes 0-15 hold K = kk+[0..15],
// lanes 16-31 hold K = kk+[16..31]  -> two contiguous 16B loads/lane.
// B fragment (32x16, col = n + lane&15): lanes 0-15 hold K = kk+[0..15],
// lanes 16-31 K = kk+[16..31] -> one contiguous 32B load/lane from h2 row.
// ---------------------------------------------------------------------------
__global__ void __launch_bounds__(128) gemm_score_kernel(
    const __bf16* __restrict__ A, const __bf16* __restrict__ B,
    const float* __restrict__ c2, float* __restrict__ score) {
  const int wave   = threadIdx.x >> 5;
  const int lane   = threadIdx.x & 31;
  const int lo16   = lane & 15;
  const int hiHalf = lane >> 4;                 // 0 or 1 (wave32)
  const int m0 = blockIdx.y * 128 + (wave >> 1) * 64;
  const int n0 = blockIdx.x * 128 + (wave & 1) * 64;

  v8f acc[4][4] = {};

  const __bf16* pa = A + (size_t)(m0 + lo16) * DDIM + hiHalf * 16;
  const __bf16* pb = B + (size_t)(n0 + lo16) * DDIM + hiHalf * 16;

  for (int kk = 0; kk < DDIM; kk += 32) {
    union { v16bf v; v8bf h[2]; } af[4];
    v16bf bf[4];
#pragma unroll
    for (int i = 0; i < 4; ++i) {
      const __bf16* p = pa + (size_t)i * 16 * DDIM + kk;
      af[i].h[0] = *(const v8bf*)(p);
      af[i].h[1] = *(const v8bf*)(p + 8);
    }
#pragma unroll
    for (int j = 0; j < 4; ++j)
      bf[j] = *(const v16bf*)(pb + (size_t)j * 16 * DDIM + kk);
    __builtin_prefetch(pa + kk + 128, 0, 1);   // global_prefetch_b8
    __builtin_prefetch(pb + kk + 128, 0, 1);
#pragma unroll
    for (int i = 0; i < 4; ++i)
#pragma unroll
      for (int j = 0; j < 4; ++j)
        acc[i][j] = __builtin_amdgcn_wmma_f32_16x16x32_bf16(
            /*neg_a=*/false, af[i].v, /*neg_b=*/false, bf[j],
            /*c_mod=*/(short)0, acc[i][j], /*reuse_a=*/false, /*reuse_b=*/false);
  }

  // Epilogue: C/D layout -> row = m0 + 16*i + r + 8*hiHalf, col = n0+16*j+lo16.
#pragma unroll
  for (int j = 0; j < 4; ++j) {
    int col = n0 + 16 * j + lo16;
    float c2v = c2[col];
#pragma unroll
    for (int i = 0; i < 4; ++i) {
      int rbase = m0 + 16 * i + hiHalf * 8;
#pragma unroll
      for (int r = 0; r < 8; ++r)
        score[(size_t)(rbase + r) * BS + col] = c2v - 2.0f * acc[i][j][r];
    }
  }
}

// ---------------------------------------------------------------------------
// Kernel 3: per-row hard mining. Block = 256 threads per row.
// pos: argmax of score over same-label cols; neg: argmin over diff-label.
// First-index tie-breaking to match jnp.argmax/argmin.
// ---------------------------------------------------------------------------
__global__ void __launch_bounds__(256) hardmine_kernel(
    const float* __restrict__ score, const int* __restrict__ labels1,
    const int* __restrict__ labels2, int* __restrict__ pos_idx,
    int* __restrict__ neg_idx, int* __restrict__ valid) {
  const int i = blockIdx.x;
  const int l1 = labels1[i];
  const float* row = score + (size_t)i * BS;

  float bp = -INFINITY; int bpi = 0x7fffffff; int hp = 0;
  float bn =  INFINITY; int bni = 0x7fffffff; int hn = 0;
  for (int j = threadIdx.x; j < BS; j += 256) {
    float v = row[j];
    if (labels2[j] == l1) {
      hp = 1;
      if (v > bp || (v == bp && j < bpi)) { bp = v; bpi = j; }
    } else {
      hn = 1;
      if (v < bn || (v == bn && j < bni)) { bn = v; bni = j; }
    }
  }

  __shared__ float sv[256]; __shared__ int si[256]; __shared__ int sf[256];

  // positive reduction (max, first index wins ties)
  sv[threadIdx.x] = bp; si[threadIdx.x] = bpi; sf[threadIdx.x] = hp;
  __syncthreads();
  for (int off = 128; off > 0; off >>= 1) {
    if (threadIdx.x < off) {
      float ov = sv[threadIdx.x + off]; int oi = si[threadIdx.x + off];
      if (ov > sv[threadIdx.x] || (ov == sv[threadIdx.x] && oi < si[threadIdx.x])) {
        sv[threadIdx.x] = ov; si[threadIdx.x] = oi;
      }
      sf[threadIdx.x] |= sf[threadIdx.x + off];
    }
    __syncthreads();
  }
  int posi = 0, hasp = 0;
  if (threadIdx.x == 0) {
    posi = (si[0] == 0x7fffffff) ? 0 : si[0];
    hasp = sf[0];
  }
  __syncthreads();

  // negative reduction (min, first index wins ties)
  sv[threadIdx.x] = bn; si[threadIdx.x] = bni; sf[threadIdx.x] = hn;
  __syncthreads();
  for (int off = 128; off > 0; off >>= 1) {
    if (threadIdx.x < off) {
      float ov = sv[threadIdx.x + off]; int oi = si[threadIdx.x + off];
      if (ov < sv[threadIdx.x] || (ov == sv[threadIdx.x] && oi < si[threadIdx.x])) {
        sv[threadIdx.x] = ov; si[threadIdx.x] = oi;
      }
      sf[threadIdx.x] |= sf[threadIdx.x + off];
    }
    __syncthreads();
  }
  if (threadIdx.x == 0) {
    pos_idx[i] = posi;
    neg_idx[i] = (si[0] == 0x7fffffff) ? 0 : si[0];
    valid[i]   = hasp & sf[0];
  }
}

// ---------------------------------------------------------------------------
// Kernel 4a: zero the accumulator (loss_sum, cnt).
// ---------------------------------------------------------------------------
__global__ void zero_kernel(float* acc) { acc[0] = 0.0f; acc[1] = 0.0f; }

// ---------------------------------------------------------------------------
// Kernel 4b: exact fp32 triplet distances + global accumulation.
// ---------------------------------------------------------------------------
__global__ void __launch_bounds__(256) triplet_kernel(
    const float* __restrict__ h1, const float* __restrict__ h2,
    const int* __restrict__ pos_idx, const int* __restrict__ neg_idx,
    const int* __restrict__ valid, float* __restrict__ acc) {
  const int i = blockIdx.x;
  const float* a = h1 + (size_t)i * DDIM;
  const float* p = h2 + (size_t)pos_idx[i] * DDIM;
  const float* n = h2 + (size_t)neg_idx[i] * DDIM;

  float sap = 0.0f, san = 0.0f;
  for (int k = threadIdx.x; k < DDIM; k += 256) {
    float av = a[k];
    float dp = av - p[k] + EPS; sap += dp * dp;
    float dn = av - n[k] + EPS; san += dn * dn;
  }
  __shared__ float r0[256], r1[256];
  r0[threadIdx.x] = sap; r1[threadIdx.x] = san;
  __syncthreads();
  for (int off = 128; off > 0; off >>= 1) {
    if (threadIdx.x < off) {
      r0[threadIdx.x] += r0[threadIdx.x + off];
      r1[threadIdx.x] += r1[threadIdx.x + off];
    }
    __syncthreads();
  }
  if (threadIdx.x == 0 && valid[i]) {
    float per = fmaxf(sqrtf(r0[0]) - sqrtf(r1[0]), 0.0f);
    atomicAdd(&acc[0], per);
    atomicAdd(&acc[1], 1.0f);
  }
}

// ---------------------------------------------------------------------------
// Kernel 5: final scalar.
// ---------------------------------------------------------------------------
__global__ void finalize_kernel(const float* __restrict__ acc, float* __restrict__ out) {
  float cnt = acc[1];
  out[0] = (cnt > 0.0f) ? (acc[0] / fmaxf(cnt, 1.0f)) : 0.0f;
}

extern "C" void kernel_launch(void* const* d_in, const int* in_sizes, int n_in,
                              void* d_out, int out_size, void* d_ws, size_t ws_size,
                              hipStream_t stream) {
  const float* h1 = (const float*)d_in[0];
  const float* h2 = (const float*)d_in[1];
  const int* labels1 = (const int*)d_in[2];
  const int* labels2 = (const int*)d_in[3];
  float* out = (float*)d_out;

  // Workspace layout (all 256B aligned):
  char* ws = (char*)d_ws;
  constexpr size_t SZ_BF = (size_t)BS * DDIM * sizeof(__bf16);   // 16 MB each
  constexpr size_t SZ_SC = (size_t)BS * BS * sizeof(float);      // 64 MB
  __bf16* b1    = (__bf16*)(ws);
  __bf16* b2    = (__bf16*)(ws + SZ_BF);
  float*  c2    = (float*) (ws + 2 * SZ_BF);
  float*  score = (float*) (ws + 2 * SZ_BF + 16384);
  int* pos_idx  = (int*)   (ws + 2 * SZ_BF + 16384 + SZ_SC);
  int* neg_idx  = (int*)   (ws + 2 * SZ_BF + 16384 + SZ_SC + 16384);
  int* valid    = (int*)   (ws + 2 * SZ_BF + 16384 + SZ_SC + 32768);
  float* acc    = (float*) (ws + 2 * SZ_BF + 16384 + SZ_SC + 49152);

  prep_kernel<<<dim3(2 * BS), dim3(256), 0, stream>>>(h1, h2, b1, b2, c2);
  gemm_score_kernel<<<dim3(BS / 128, BS / 128), dim3(128), 0, stream>>>(b1, b2, c2, score);
  hardmine_kernel<<<dim3(BS), dim3(256), 0, stream>>>(score, labels1, labels2,
                                                      pos_idx, neg_idx, valid);
  zero_kernel<<<dim3(1), dim3(1), 0, stream>>>(acc);
  triplet_kernel<<<dim3(BS), dim3(256), 0, stream>>>(h1, h2, pos_idx, neg_idx, valid, acc);
  finalize_kernel<<<dim3(1), dim3(1), 0, stream>>>(acc, out);
}
